// Sequence_38199439130725
// MI455X (gfx1250) — compile-verified, matrix operand order checked by reading, not run on hardware
//
#include <hip/hip_runtime.h>

// ---------------------------------------------------------------------------
// 2-layer GRU (H=256) + linear head, persistent WMMA kernel for gfx1250.
//   B=2048 batch split into 64 workgroups x 32 rows; T steps looped in-kernel.
//   Weights repacked once into bf16 WMMA-B fragment order in d_ws (L2-resident).
//   v_wmma_f32_16x16x32_bf16 for all GEMMs; f32 hidden state in registers.
// ---------------------------------------------------------------------------

typedef __attribute__((ext_vector_type(16))) __bf16 v16bf;
typedef __attribute__((ext_vector_type(8)))  float  v8f;

typedef unsigned short ushort_t;
typedef unsigned int   uint_t;

#define HDIM   256
#define GDIM   768            // 3*H gate rows
#define MT     32             // batch rows per workgroup
#define NTHR   256            // 8 waves

// ---- workspace layout (bytes) ----
// packed tiles: 32x16 (KxN) bf16, 1024B each, lane-major (lane*32B)
#define OFF_WIH1P 0u                       // 48 tiles (K padded 8->32)
#define OFF_WHH1P (OFF_WIH1P + 48u*1024u)  // 384 tiles
#define OFF_WIH2P (OFF_WHH1P + 384u*1024u)
#define OFF_WHH2P (OFF_WIH2P + 384u*1024u)
#define OFF_LINP  (OFF_WHH2P + 384u*1024u) // 8 tiles (rows padded 5->16)
#define OFF_BIH1  (OFF_LINP  + 8u*1024u)
#define OFF_BHH1  (OFF_BIH1 + 768u*4u)
#define OFF_BIH2  (OFF_BHH1 + 768u*4u)
#define OFF_BHH2  (OFF_BIH2 + 768u*4u)
#define OFF_LINB  (OFF_BHH2 + 768u*4u)     // 16 floats (padded)

__device__ __forceinline__ ushort_t f2bf(float f) {
    uint_t u = __float_as_uint(f);
    u += 0x7FFFu + ((u >> 16) & 1u);       // round-to-nearest-even
    return (ushort_t)(u >> 16);
}
__device__ __forceinline__ float sigm(float x) { return 1.0f / (1.0f + __expf(-x)); }
__device__ __forceinline__ float tanh_(float x) {
    float e = __expf(-2.0f * x);
    return (1.0f - e) / (1.0f + e);
}

union FragBF { v16bf v; uint4 q[2]; };

// A fragment (16x32 bf16, M x K) from row-major bf16 LDS.
// Lane L: M = L%16, K-chunks of 8 at kbase and kbase+16 (kbase = kt*32 + (L/16)*8).
__device__ __forceinline__ FragBF load_a(const ushort_t* base, int mrow, int stride, int kbase) {
    FragBF f;
    f.q[0] = *(const uint4*)(base + mrow * stride + kbase);
    f.q[1] = *(const uint4*)(base + mrow * stride + kbase + 16);
    return f;
}
// B fragment (32x16 bf16, K x N) from packed global blob: 1024B/tile, 32B/lane.
__device__ __forceinline__ FragBF load_b(const char* ws, uint_t off, int tile, int lane) {
    FragBF f;
    const uint4* p = (const uint4*)(ws + off + (size_t)tile * 1024u + lane * 32);
    f.q[0] = p[0];
    f.q[1] = p[1];
    return f;
}
__device__ __forceinline__ v8f wmma(v8f c, FragBF a, FragBF b) {
    return __builtin_amdgcn_wmma_f32_16x16x32_bf16(false, a.v, false, b.v,
                                                   (short)0, c, false, false);
}

// ---------------------------------------------------------------------------
// Weight packer: one block per 32x16 tile; lane L emits its 16 bf16 values
// in WMMA-B register order (mirrors the 16-bit A layout with M<->N swapped):
//   n = nt*16 + L%16 ; K positions  koff+p (p<8) and koff+16+(p-8), koff=(L/16)*8
// Source W is row-major [rows x cols]; B[k][n] = W[n][k] (we compute h @ W^T).
// ---------------------------------------------------------------------------
__global__ void pack_weights(const float* __restrict__ wih1, const float* __restrict__ whh1,
                             const float* __restrict__ wih2, const float* __restrict__ whh2,
                             const float* __restrict__ linw, char* __restrict__ ws) {
    int blk = blockIdx.x, lane = threadIdx.x;
    const float* src; uint_t dsto; int tile, nt, kt, rows, cols;
    if (blk < 48)        { src = wih1; dsto = OFF_WIH1P; tile = blk;        nt = tile;     kt = 0;        rows = GDIM; cols = 8; }
    else if (blk < 432)  { src = whh1; dsto = OFF_WHH1P; tile = blk - 48;   nt = tile >> 3; kt = tile & 7; rows = GDIM; cols = HDIM; }
    else if (blk < 816)  { src = wih2; dsto = OFF_WIH2P; tile = blk - 432;  nt = tile >> 3; kt = tile & 7; rows = GDIM; cols = HDIM; }
    else if (blk < 1200) { src = whh2; dsto = OFF_WHH2P; tile = blk - 816;  nt = tile >> 3; kt = tile & 7; rows = GDIM; cols = HDIM; }
    else                 { src = linw; dsto = OFF_LINP;  tile = blk - 1200; nt = 0;        kt = tile;     rows = 5;    cols = HDIM; }

    int n    = nt * 16 + (lane & 15);
    int koff = (lane >> 4) * 8;
    ushort_t vals[16];
#pragma unroll
    for (int p = 0; p < 16; ++p) {
        int kl = (p < 8) ? (koff + p) : (koff + 16 + (p - 8));
        int k  = kt * 32 + kl;
        float v = (n < rows && k < cols) ? src[n * cols + k] : 0.0f;
        vals[p] = f2bf(v);
    }
    uint4* o = (uint4*)(ws + dsto + (size_t)tile * 1024u + lane * 32);
    o[0] = *(uint4*)&vals[0];
    o[1] = *(uint4*)&vals[8];
}

__global__ void pack_bias(const float* __restrict__ bih1, const float* __restrict__ bhh1,
                          const float* __restrict__ bih2, const float* __restrict__ bhh2,
                          const float* __restrict__ linb, char* __restrict__ ws) {
    int i = blockIdx.x * blockDim.x + threadIdx.x;
    if (i < GDIM) {
        ((float*)(ws + OFF_BIH1))[i] = bih1[i];
        ((float*)(ws + OFF_BHH1))[i] = bhh1[i];
        ((float*)(ws + OFF_BIH2))[i] = bih2[i];
        ((float*)(ws + OFF_BHH2))[i] = bhh2[i];
    }
    if (i < 16) ((float*)(ws + OFF_LINB))[i] = (i < 5) ? linb[i] : 0.0f;
}

// ---------------------------------------------------------------------------
// Persistent GRU kernel. One workgroup = 32 batch rows. Wave w owns hidden
// columns [32w, 32w+32) (two 16-col tiles) for BOTH 16-row batch tiles.
// C-layout: vgpr i of lane L holds (m = i + 8*(L/16), n = L%16).
// ---------------------------------------------------------------------------
__global__ void __launch_bounds__(NTHR)
gru_persistent(const float* __restrict__ rnn, const float* __restrict__ out0,
               const float* __restrict__ h01, const float* __restrict__ h02,
               const char* __restrict__ ws, float* __restrict__ out,
               int T, int Btot) {
    __shared__ ushort_t h1b[MT * HDIM];   // bf16 mirror of h1 (WMMA A source)
    __shared__ ushort_t h2b[MT * HDIM];
    __shared__ ushort_t xb[MT * 32];      // x = [out(5), u(3)] padded K->32
    __shared__ float bi1[GDIM], bh1[GDIM], bi2[GDIM], bh2[GDIM], lb[16];

    const int tid   = threadIdx.x;
    const int w     = tid >> 5;
    const int lane  = tid & 31;
    const int lrow  = lane & 15;     // N-column within 16-wide tile
    const int lhalf = lane >> 4;
    const int base  = blockIdx.x * MT;
    const int c0    = w * 32;        // this wave's hidden-column base

    // ---- prologue: biases + initial state into LDS / registers ----
    for (int i = tid; i < GDIM; i += NTHR) {
        bi1[i] = ((const float*)(ws + OFF_BIH1))[i];
        bh1[i] = ((const float*)(ws + OFF_BHH1))[i];
        bi2[i] = ((const float*)(ws + OFF_BIH2))[i];
        bh2[i] = ((const float*)(ws + OFF_BHH2))[i];
    }
    if (tid < 16) lb[tid] = ((const float*)(ws + OFF_LINB))[tid];

    for (int i = tid; i < MT * HDIM; i += NTHR) {
        int m = i >> 8, c = i & 255;
        h1b[i] = f2bf(h01[(size_t)(base + m) * HDIM + c]);
        h2b[i] = f2bf(h02[(size_t)(base + m) * HDIM + c]);
    }
    for (int i = tid; i < MT * 32; i += NTHR) {
        int m = i >> 5, c = i & 31;
        float v = 0.0f;
        if (c < 5)      v = out0[(size_t)(base + m) * 5 + c];
        else if (c < 8) v = rnn[(size_t)(base + m) * 4 + (c - 5)];   // t=0
        xb[i] = f2bf(v);
    }
    // f32 hidden state in registers at this lane's C-layout positions
    float h1o[2][2][8], h2o[2][2][8];
#pragma unroll
    for (int s = 0; s < 2; ++s)
#pragma unroll
        for (int mt = 0; mt < 2; ++mt)
#pragma unroll
            for (int i = 0; i < 8; ++i) {
                int m = mt * 16 + lhalf * 8 + i;
                int j = c0 + s * 16 + lrow;
                h1o[s][mt][i] = h01[(size_t)(base + m) * HDIM + j];
                h2o[s][mt][i] = h02[(size_t)(base + m) * HDIM + j];
            }
    // warm L2 with the recurrent weights
    for (uint_t off = (uint_t)tid * 128u; off < 3u * 384u * 1024u; off += NTHR * 128u)
        __builtin_prefetch(ws + OFF_WHH1P + off, 0, 1);
    __syncthreads();

    // =================== time loop ===================
    for (int t = 0; t < T; ++t) {
        // ---------- layer 1: gates = x@wih1^T + h1@whh1^T ----------
#pragma unroll 1
        for (int s = 0; s < 2; ++s) {
            const int ht = w * 2 + s;
            const int j  = c0 + s * 16 + lrow;
            v8f Cr[2], Cz[2], Ci[2], Ch[2];
            {
                float br = bi1[j] + bh1[j], bz = bi1[HDIM + j] + bh1[HDIM + j];
                float bn = bi1[2 * HDIM + j], bh = bh1[2 * HDIM + j];
#pragma unroll
                for (int i = 0; i < 8; ++i) {
                    Cr[0][i] = br; Cr[1][i] = br; Cz[0][i] = bz; Cz[1][i] = bz;
                    Ci[0][i] = bn; Ci[1][i] = bn; Ch[0][i] = bh; Ch[1][i] = bh;
                }
            }
            {   // x contribution: single padded K-tile
                FragBF a0 = load_a(xb, lrow, 32, lhalf * 8);
                FragBF a1 = load_a(xb, 16 + lrow, 32, lhalf * 8);
                FragBF br_ = load_b(ws, OFF_WIH1P, ht, lane);
                FragBF bz_ = load_b(ws, OFF_WIH1P, 16 + ht, lane);
                FragBF bn_ = load_b(ws, OFF_WIH1P, 32 + ht, lane);
                Cr[0] = wmma(Cr[0], a0, br_); Cr[1] = wmma(Cr[1], a1, br_);
                Cz[0] = wmma(Cz[0], a0, bz_); Cz[1] = wmma(Cz[1], a1, bz_);
                Ci[0] = wmma(Ci[0], a0, bn_); Ci[1] = wmma(Ci[1], a1, bn_);
            }
#pragma unroll 2
            for (int kt = 0; kt < 8; ++kt) {
                FragBF a0 = load_a(h1b, lrow, HDIM, kt * 32 + lhalf * 8);
                FragBF a1 = load_a(h1b, 16 + lrow, HDIM, kt * 32 + lhalf * 8);
                FragBF wr = load_b(ws, OFF_WHH1P, ht * 8 + kt, lane);
                FragBF wz = load_b(ws, OFF_WHH1P, (16 + ht) * 8 + kt, lane);
                FragBF wn = load_b(ws, OFF_WHH1P, (32 + ht) * 8 + kt, lane);
                Cr[0] = wmma(Cr[0], a0, wr); Cr[1] = wmma(Cr[1], a1, wr);
                Cz[0] = wmma(Cz[0], a0, wz); Cz[1] = wmma(Cz[1], a1, wz);
                Ch[0] = wmma(Ch[0], a0, wn); Ch[1] = wmma(Ch[1], a1, wn);
            }
#pragma unroll
            for (int mt = 0; mt < 2; ++mt)
#pragma unroll
                for (int i = 0; i < 8; ++i) {
                    float r = sigm(Cr[mt][i]);
                    float z = sigm(Cz[mt][i]);
                    float n = tanh_(Ci[mt][i] + r * Ch[mt][i]);
                    h1o[s][mt][i] = (1.0f - z) * n + z * h1o[s][mt][i];
                }
        }
        __syncthreads();                       // all reads of h1b/xb done
#pragma unroll
        for (int s = 0; s < 2; ++s)
#pragma unroll
            for (int mt = 0; mt < 2; ++mt)
#pragma unroll
                for (int i = 0; i < 8; ++i) {
                    int m = mt * 16 + lhalf * 8 + i;
                    h1b[m * HDIM + c0 + s * 16 + lrow] = f2bf(h1o[s][mt][i]);
                }
        __syncthreads();                       // h1b (new) published

        // ---------- layer 2: gates = h1@wih2^T + h2@whh2^T ----------
#pragma unroll 1
        for (int s = 0; s < 2; ++s) {
            const int ht = w * 2 + s;
            const int j  = c0 + s * 16 + lrow;
            v8f Cr[2], Cz[2], Ci[2], Ch[2];
            {
                float br = bi2[j] + bh2[j], bz = bi2[HDIM + j] + bh2[HDIM + j];
                float bn = bi2[2 * HDIM + j], bh = bh2[2 * HDIM + j];
#pragma unroll
                for (int i = 0; i < 8; ++i) {
                    Cr[0][i] = br; Cr[1][i] = br; Cz[0][i] = bz; Cz[1][i] = bz;
                    Ci[0][i] = bn; Ci[1][i] = bn; Ch[0][i] = bh; Ch[1][i] = bh;
                }
            }
#pragma unroll 2
            for (int kt = 0; kt < 8; ++kt) {
                FragBF a0 = load_a(h1b, lrow, HDIM, kt * 32 + lhalf * 8);
                FragBF a1 = load_a(h1b, 16 + lrow, HDIM, kt * 32 + lhalf * 8);
                FragBF g0 = load_a(h2b, lrow, HDIM, kt * 32 + lhalf * 8);
                FragBF g1 = load_a(h2b, 16 + lrow, HDIM, kt * 32 + lhalf * 8);
                FragBF ir = load_b(ws, OFF_WIH2P, ht * 8 + kt, lane);
                FragBF iz = load_b(ws, OFF_WIH2P, (16 + ht) * 8 + kt, lane);
                FragBF in_ = load_b(ws, OFF_WIH2P, (32 + ht) * 8 + kt, lane);
                FragBF hr = load_b(ws, OFF_WHH2P, ht * 8 + kt, lane);
                FragBF hz = load_b(ws, OFF_WHH2P, (16 + ht) * 8 + kt, lane);
                FragBF hn = load_b(ws, OFF_WHH2P, (32 + ht) * 8 + kt, lane);
                Cr[0] = wmma(Cr[0], a0, ir); Cr[1] = wmma(Cr[1], a1, ir);
                Cr[0] = wmma(Cr[0], g0, hr); Cr[1] = wmma(Cr[1], g1, hr);
                Cz[0] = wmma(Cz[0], a0, iz); Cz[1] = wmma(Cz[1], a1, iz);
                Cz[0] = wmma(Cz[0], g0, hz); Cz[1] = wmma(Cz[1], g1, hz);
                Ci[0] = wmma(Ci[0], a0, in_); Ci[1] = wmma(Ci[1], a1, in_);
                Ch[0] = wmma(Ch[0], g0, hn); Ch[1] = wmma(Ch[1], g1, hn);
            }
#pragma unroll
            for (int mt = 0; mt < 2; ++mt)
#pragma unroll
                for (int i = 0; i < 8; ++i) {
                    float r = sigm(Cr[mt][i]);
                    float z = sigm(Cz[mt][i]);
                    float n = tanh_(Ci[mt][i] + r * Ch[mt][i]);
                    h2o[s][mt][i] = (1.0f - z) * n + z * h2o[s][mt][i];
                }
        }
        __syncthreads();                       // all reads of h1b/h2b done
#pragma unroll
        for (int s = 0; s < 2; ++s)
#pragma unroll
            for (int mt = 0; mt < 2; ++mt)
#pragma unroll
                for (int i = 0; i < 8; ++i) {
                    int m = mt * 16 + lhalf * 8 + i;
                    h2b[m * HDIM + c0 + s * 16 + lrow] = f2bf(h2o[s][mt][i]);
                }
        __syncthreads();                       // h2b (new) published

        // ---------- head: out = h2@lin_w^T + lin_b ; build next x ----------
        if (w < 2) {                           // wave w -> batch m-tile w
            v8f Co;
#pragma unroll
            for (int i = 0; i < 8; ++i) Co[i] = lb[lrow];
#pragma unroll 2
            for (int kt = 0; kt < 8; ++kt) {
                FragBF a  = load_a(h2b, w * 16 + lrow, HDIM, kt * 32 + lhalf * 8);
                FragBF bl = load_b(ws, OFF_LINP, kt, lane);
                Co = wmma(Co, a, bl);
            }
            if (lrow < 5) {
#pragma unroll
                for (int i = 0; i < 8; ++i) {
                    int m = w * 16 + lhalf * 8 + i;
                    out[(size_t)(base + m) * T * 5 + (size_t)t * 5 + lrow] = Co[i];
                    xb[m * 32 + lrow] = f2bf(Co[i]);
                }
            }
        }
        if (t + 1 < T && tid >= 96 && tid < 192) {   // idle waves fetch u_{t+1}
            int idx = tid - 96, m = idx / 3, c = idx % 3;
            xb[m * 32 + 5 + c] =
                f2bf(rnn[(size_t)(t + 1) * Btot * 4 + (size_t)(base + m) * 4 + c]);
        }
        __syncthreads();                       // xb ready for next step
    }
}

extern "C" void kernel_launch(void* const* d_in, const int* in_sizes, int n_in,
                              void* d_out, int out_size, void* d_ws, size_t ws_size,
                              hipStream_t stream) {
    const float* rnn  = (const float*)d_in[1];
    const float* out0 = (const float*)d_in[2];
    const float* h01  = (const float*)d_in[3];
    const float* h02  = (const float*)d_in[4];
    const float* wih1 = (const float*)d_in[5];
    const float* whh1 = (const float*)d_in[6];
    const float* bih1 = (const float*)d_in[7];
    const float* bhh1 = (const float*)d_in[8];
    const float* wih2 = (const float*)d_in[9];
    const float* whh2 = (const float*)d_in[10];
    const float* bih2 = (const float*)d_in[11];
    const float* bhh2 = (const float*)d_in[12];
    const float* linw = (const float*)d_in[13];
    const float* linb = (const float*)d_in[14];

    const int Btot = in_sizes[2] / 5;                 // out0 is (B, 5)
    const int T    = out_size / (Btot * 5);           // d_out is (B, T, 5)
    char* ws = (char*)d_ws;

    pack_weights<<<1208, 32, 0, stream>>>(wih1, whh1, wih2, whh2, linw, ws);
    pack_bias<<<3, 256, 0, stream>>>(bih1, bhh1, bih2, bhh2, linb, ws);
    gru_persistent<<<Btot / MT, NTHR, 0, stream>>>(rnn, out0, h01, h02, ws,
                                                   (float*)d_out, T, Btot);
}
